// Block_28741921145208
// MI455X (gfx1250) — matrix operand order repaired and checked by reading, not compiled, not run on hardware
//
#include <hip/hip_runtime.h>

typedef unsigned short u16;
typedef __bf16 bf16_t;
typedef bf16_t bf16x16 __attribute__((ext_vector_type(16)));
typedef float f32x8 __attribute__((ext_vector_type(8)));

#define SEQ 2048
#define CDIM 384
#define NHEAD 6
#define HDIM 64
#define HID 1536
#define NTOK 8192   // B*SEQ
#define LDSW 72     // LDS row stride in u16: 144B = 16B-aligned, conflict-free (gcd(36,64)=4)

union Frag { bf16x16 v; uint4 q[2]; };

__device__ __forceinline__ u16 f2bf(float f) {
    unsigned u = __builtin_bit_cast(unsigned, f);
    u += 0x7FFFu + ((u >> 16) & 1u);
    return (u16)(u >> 16);
}

__device__ __forceinline__ f32x8 wmma_bf16(bf16x16 a, bf16x16 b, f32x8 c) {
    return __builtin_amdgcn_wmma_f32_16x16x32_bf16(false, a, false, b, (short)0, c, false, false);
}

__device__ __forceinline__ f32x8 zero8() {
    f32x8 z = {0.f, 0.f, 0.f, 0.f, 0.f, 0.f, 0.f, 0.f};
    return z;
}

// Async global->LDS 16B copy (per lane), tracked by ASYNCcnt.
// lds_off: LDS byte address (low 32 bits of flat shared pointer; LDS aperture
// is encoded entirely in ADDR[63:32] per the CDNA5 aperture check).
#define ASYNC_LD16(ldsoff, gaddr, OFFSTR)                                      \
    asm volatile("global_load_async_to_lds_b128 %0, %1, off offset:" OFFSTR    \
                 ::"v"(ldsoff), "v"(gaddr)                                     \
                 : "memory")

__device__ __forceinline__ void issue_slab(const u16* Ag, const u16* Bg,
                                           unsigned aoff, unsigned boff) {
    unsigned long long a = (unsigned long long)(uintptr_t)Ag;
    unsigned long long b = (unsigned long long)(uintptr_t)Bg;
    ASYNC_LD16(aoff, a, "0");
    ASYNC_LD16(aoff, a, "16");
    ASYNC_LD16(aoff, a, "32");
    ASYNC_LD16(aoff, a, "48");
    ASYNC_LD16(boff, b, "0");
    ASYNC_LD16(boff, b, "16");
    ASYNC_LD16(boff, b, "32");
    ASYNC_LD16(boff, b, "48");
}

// ---------------- fp32 -> bf16 convert ----------------
__global__ __launch_bounds__(256) void cvt_kernel(const float* __restrict__ in,
                                                  u16* __restrict__ out, int n) {
    for (int i = blockIdx.x * 256 + threadIdx.x; i < n; i += gridDim.x * 256)
        out[i] = f2bf(in[i]);
}

// fp32 [K,N] -> bf16 transposed [N,K] (one-time weight prep; makes GEMM B staging
// a straight row copy so both operands can use async-to-LDS)
__global__ __launch_bounds__(256) void cvt_t_kernel(const float* __restrict__ in,
                                                    u16* __restrict__ out,
                                                    int K, int N) {
    int total = K * N;
    for (int i = blockIdx.x * 256 + threadIdx.x; i < total; i += gridDim.x * 256) {
        int k = i / N;
        int n = i - k * N;
        out[(size_t)n * K + k] = f2bf(in[i]);
    }
}

// ---------------- LayerNorm: one wave per token ----------------
__global__ __launch_bounds__(256) void ln_kernel(const float* __restrict__ x,
                                                 const float* __restrict__ g,
                                                 const float* __restrict__ b,
                                                 u16* __restrict__ out) {
    int lane = threadIdx.x & 31;
    int wave = threadIdx.x >> 5;
    int token = blockIdx.x * 8 + wave;
    const float* xr = x + (size_t)token * CDIM;

    float v[12];
    float sum = 0.f;
#pragma unroll
    for (int i = 0; i < 12; i++) { v[i] = xr[lane + 32 * i]; sum += v[i]; }
#pragma unroll
    for (int m = 1; m < 32; m <<= 1) sum += __shfl_xor(sum, m, 32);
    float mean = sum * (1.0f / CDIM);

    float sq = 0.f;
#pragma unroll
    for (int i = 0; i < 12; i++) { float d = v[i] - mean; sq += d * d; }
#pragma unroll
    for (int m = 1; m < 32; m <<= 1) sq += __shfl_xor(sq, m, 32);
    float rstd = rsqrtf(sq * (1.0f / CDIM) + 1e-5f);

#pragma unroll
    for (int i = 0; i < 12; i++) {
        int c = lane + 32 * i;
        out[(size_t)token * CDIM + c] = f2bf((v[i] - mean) * rstd * g[c] + b[c]);
    }
}

// ---------------- bf16 GEMM, 128x128 block tile, K-step 64, async double-buffer ----
// A: [M,K] bf16 row-major, BwT: [N,K] bf16 row-major (pre-transposed weights).
// EPI 0: qkv scatter (bias; q gets *0.125; write q/k/vt bf16)
// EPI 1: bias + residual -> fp32 outf
// EPI 2: bias + exact GELU -> bf16 outh
template <int EPI>
__global__ __launch_bounds__(256) void gemm_bf16(
    const u16* __restrict__ A, const u16* __restrict__ BwT,
    const float* __restrict__ bias, const float* __restrict__ resid,
    float* __restrict__ outf, u16* __restrict__ outh,
    u16* __restrict__ qo, u16* __restrict__ ko, u16* __restrict__ vto,
    int M, int N, int K) {
    __shared__ u16 As[2][128][LDSW];   // [buf][m][k 0..63]
    __shared__ u16 Bs[2][128][LDSW];   // [buf][n][k 0..63]

    const int lane = threadIdx.x & 31;
    const int wave = threadIdx.x >> 5;
    const int wm = wave >> 1;          // 0..3 -> 32-row strip
    const int wn = wave & 1;           // 0..1 -> 64-col strip
    const int l15 = lane & 15;
    const int hf = lane >> 4;
    const int m0 = blockIdx.y * 128;
    const int n0 = blockIdx.x * 128;

    // staging-thread roles: 2 threads per row, 64B (4x16B) each
    const int srow = threadIdx.x >> 1;          // 0..127
    const int shalf = threadIdx.x & 1;          // 0..1

    const u16* aBase = A + (size_t)(m0 + srow) * K + 32 * shalf;
    const u16* bBase = BwT + (size_t)(n0 + srow) * K + 32 * shalf;
    unsigned aoff[2], boff[2];
    aoff[0] = (unsigned)(uintptr_t)&As[0][srow][32 * shalf];
    aoff[1] = (unsigned)(uintptr_t)&As[1][srow][32 * shalf];
    boff[0] = (unsigned)(uintptr_t)&Bs[0][srow][32 * shalf];
    boff[1] = (unsigned)(uintptr_t)&Bs[1][srow][32 * shalf];

    f32x8 acc[2][4];
#pragma unroll
    for (int i = 0; i < 2; i++)
#pragma unroll
        for (int j = 0; j < 4; j++) acc[i][j] = zero8();

    issue_slab(aBase, bBase, aoff[0], boff[0]);
    for (int k0 = 0; k0 < K; k0 += 64) {
        int buf = (k0 >> 6) & 1;
        asm volatile("s_wait_asynccnt 0" ::: "memory");   // own slab landed in LDS
        __syncthreads();                                   // everyone's slab landed
        if (k0 + 64 < K)                                   // background-load next slab
            issue_slab(aBase + k0 + 64, bBase + k0 + 64, aoff[buf ^ 1], boff[buf ^ 1]);

#pragma unroll
        for (int ks = 0; ks < 64; ks += 32) {
            Frag a[2], b[4];
#pragma unroll
            for (int mi = 0; mi < 2; mi++) {
                int row = 32 * wm + 16 * mi + l15;
                a[mi].q[0] = *reinterpret_cast<const uint4*>(&As[buf][row][ks + 8 * hf]);
                a[mi].q[1] = *reinterpret_cast<const uint4*>(&As[buf][row][ks + 16 + 8 * hf]);
            }
#pragma unroll
            for (int ni = 0; ni < 4; ni++) {
                int col = 64 * wn + 16 * ni + l15;
                b[ni].q[0] = *reinterpret_cast<const uint4*>(&Bs[buf][col][ks + 8 * hf]);
                b[ni].q[1] = *reinterpret_cast<const uint4*>(&Bs[buf][col][ks + 16 + 8 * hf]);
            }
#pragma unroll
            for (int mi = 0; mi < 2; mi++)
#pragma unroll
                for (int ni = 0; ni < 4; ni++)
                    acc[mi][ni] = wmma_bf16(a[mi].v, b[ni].v, acc[mi][ni]);
        }
        __syncthreads();   // compute done before next slab may overwrite this buffer
    }

    // epilogue: element (mi,ni,r) -> row = r + 8*hf, col = lane&15 in its 16x16 tile
#pragma unroll
    for (int mi = 0; mi < 2; mi++) {
#pragma unroll
        for (int ni = 0; ni < 4; ni++) {
            int col = n0 + 64 * wn + 16 * ni + l15;
            float bv = bias[col];
            int rowb = m0 + 32 * wm + 16 * mi + 8 * hf;
            if (EPI == 0) {
                int which = col / CDIM;
                int c2 = col - which * CDIM;
                int head = c2 >> 6;       // /HDIM
                int d = c2 & 63;
                float scl = (which == 0) ? 0.125f : 1.0f;   // fold attn scale into Q
#pragma unroll
                for (int r = 0; r < 8; r++) {
                    int row = rowb + r;
                    int bidx = row >> 11;                    // /SEQ
                    int n = row & (SEQ - 1);
                    float v = (acc[mi][ni][r] + bv) * scl;
                    u16 hv = f2bf(v);
                    size_t bhead = (size_t)bidx * NHEAD + head;
                    if (which == 0)
                        qo[(bhead * SEQ + n) * HDIM + d] = hv;
                    else if (which == 1)
                        ko[(bhead * SEQ + n) * HDIM + d] = hv;
                    else
                        vto[(bhead * HDIM + d) * SEQ + n] = hv;
                }
            } else if (EPI == 1) {
#pragma unroll
                for (int r = 0; r < 8; r++) {
                    size_t idx = (size_t)(rowb + r) * N + col;
                    outf[idx] = acc[mi][ni][r] + bv + resid[idx];
                }
            } else {
#pragma unroll
                for (int r = 0; r < 8; r++) {
                    float v = acc[mi][ni][r] + bv;
                    float gl = 0.5f * v * (1.0f + erff(v * 0.70710678118654752f));
                    outh[(size_t)(rowb + r) * N + col] = f2bf(gl);
                }
            }
        }
    }
}

// ---------------- flash attention: one wave per 16-query tile, 64-key chunks ----------------
// Q is pre-scaled by 1/sqrt(HDIM). Row-sum of P computed via WMMA against a ones matrix.
__global__ __launch_bounds__(256) void attn_kernel(const u16* __restrict__ q,
                                                   const u16* __restrict__ k,
                                                   const u16* __restrict__ vt,
                                                   u16* __restrict__ ao) {
    __shared__ u16 Ps[8][16][LDSW];  // per-wave P staging tile (16x64 bf16)

    const int lane = threadIdx.x & 31;
    const int wave = threadIdx.x >> 5;
    const int l15 = lane & 15;
    const int hf = lane >> 4;

    int tile = blockIdx.x * 8 + wave;   // 0..3071
    int bh = tile >> 7;                 // b*NHEAD + head
    int qt = tile & 127;
    int bidx = bh / NHEAD;
    int head = bh % NHEAD;

    const u16* qbase = q + ((size_t)bh * SEQ + qt * 16) * HDIM;
    const u16* kbase = k + (size_t)bh * SEQ * HDIM;
    const u16* vbase = vt + (size_t)bh * HDIM * SEQ;

    // Q fragments: row = lane&15, d chunks per A-layout
    Frag qf[2];
    {
        const u16* qr = qbase + l15 * HDIM;
        qf[0].q[0] = *reinterpret_cast<const uint4*>(qr + 8 * hf);
        qf[0].q[1] = *reinterpret_cast<const uint4*>(qr + 16 + 8 * hf);
        qf[1].q[0] = *reinterpret_cast<const uint4*>(qr + 32 + 8 * hf);
        qf[1].q[1] = *reinterpret_cast<const uint4*>(qr + 48 + 8 * hf);
    }

    Frag ones;   // bf16 1.0 everywhere: B-frag for row-sum WMMA
    {
        unsigned c = 0x3F803F80u;
        ones.q[0] = uint4{c, c, c, c};
        ones.q[1] = uint4{c, c, c, c};
    }

    float mrow[8], lrow[8];
    f32x8 o[4];
#pragma unroll
    for (int r = 0; r < 8; r++) { mrow[r] = -1e30f; lrow[r] = 0.f; }
#pragma unroll
    for (int j = 0; j < 4; j++) o[j] = zero8();

    for (int kc = 0; kc < SEQ; kc += 64) {
        f32x8 s[4];
#pragma unroll
        for (int t = 0; t < 4; t++) s[t] = zero8();
#pragma unroll
        for (int t = 0; t < 4; t++) {
            int key = kc + t * 16 + l15;
            const u16* kr = kbase + (size_t)key * HDIM;
            Frag kf0, kf1;   // B-frags: col = key, K = d (contiguous per key row)
            kf0.q[0] = *reinterpret_cast<const uint4*>(kr + 8 * hf);
            kf0.q[1] = *reinterpret_cast<const uint4*>(kr + 16 + 8 * hf);
            kf1.q[0] = *reinterpret_cast<const uint4*>(kr + 32 + 8 * hf);
            kf1.q[1] = *reinterpret_cast<const uint4*>(kr + 48 + 8 * hf);
            s[t] = wmma_bf16(qf[0].v, kf0.v, s[t]);
            s[t] = wmma_bf16(qf[1].v, kf1.v, s[t]);
        }

        // row max across 64 columns: in-lane max over 4 tiles, then 4-step butterfly
        float alpha[8];
#pragma unroll
        for (int r = 0; r < 8; r++) {
            float mx = fmaxf(fmaxf(s[0][r], s[1][r]), fmaxf(s[2][r], s[3][r]));
            mx = fmaxf(mx, __shfl_xor(mx, 1, 32));
            mx = fmaxf(mx, __shfl_xor(mx, 2, 32));
            mx = fmaxf(mx, __shfl_xor(mx, 4, 32));
            mx = fmaxf(mx, __shfl_xor(mx, 8, 32));
            float mnew = fmaxf(mrow[r], mx);
            alpha[r] = __expf(mrow[r] - mnew);
            mrow[r] = mnew;
        }

        // p = exp(s - m), staged straight to per-wave LDS tile as bf16
#pragma unroll
        for (int t = 0; t < 4; t++)
#pragma unroll
            for (int r = 0; r < 8; r++)
                Ps[wave][8 * hf + r][16 * t + l15] = f2bf(__expf(s[t][r] - mrow[r]));
        asm volatile("s_wait_dscnt 0" ::: "memory");

        Frag pf0, pf1;   // A-frags of P (16x64 -> two K=32 fragments)
        pf0.q[0] = *reinterpret_cast<const uint4*>(&Ps[wave][l15][8 * hf]);
        pf0.q[1] = *reinterpret_cast<const uint4*>(&Ps[wave][l15][16 + 8 * hf]);
        pf1.q[0] = *reinterpret_cast<const uint4*>(&Ps[wave][l15][32 + 8 * hf]);
        pf1.q[1] = *reinterpret_cast<const uint4*>(&Ps[wave][l15][48 + 8 * hf]);

        // row-sum via WMMA against ones (replaces 32 ds_bpermute per chunk)
        f32x8 rs = wmma_bf16(pf0.v, ones.v, zero8());
        rs = wmma_bf16(pf1.v, ones.v, rs);
#pragma unroll
        for (int r = 0; r < 8; r++) lrow[r] = lrow[r] * alpha[r] + rs[r];

#pragma unroll
        for (int j = 0; j < 4; j++)
#pragma unroll
            for (int r = 0; r < 8; r++) o[j][r] *= alpha[r];

        // P @ V over 64 keys
#pragma unroll
        for (int j = 0; j < 4; j++) {
            int d = 16 * j + l15;
            const u16* vr = vbase + (size_t)d * SEQ + kc;
            Frag vfA, vfB;   // B-frags: col = d, K = keys (contiguous in vt rows)
            vfA.q[0] = *reinterpret_cast<const uint4*>(vr + 8 * hf);
            vfA.q[1] = *reinterpret_cast<const uint4*>(vr + 16 + 8 * hf);
            vfB.q[0] = *reinterpret_cast<const uint4*>(vr + 32 + 8 * hf);
            vfB.q[1] = *reinterpret_cast<const uint4*>(vr + 48 + 8 * hf);
            o[j] = wmma_bf16(pf0.v, vfA.v, o[j]);
            o[j] = wmma_bf16(pf1.v, vfB.v, o[j]);
        }
    }

    // normalize and scatter to attn_out [token, head*64+d] as bf16
    float inv[8];
#pragma unroll
    for (int r = 0; r < 8; r++) inv[r] = 1.0f / lrow[r];
#pragma unroll
    for (int j = 0; j < 4; j++) {
#pragma unroll
        for (int r = 0; r < 8; r++) {
            int n = qt * 16 + 8 * hf + r;
            size_t idx = ((size_t)bidx * SEQ + n) * CDIM + head * HDIM + 16 * j + l15;
            ao[idx] = f2bf(o[j][r] * inv[r]);
        }
    }
}

extern "C" void kernel_launch(void* const* d_in, const int* in_sizes, int n_in,
                              void* d_out, int out_size, void* d_ws, size_t ws_size,
                              hipStream_t stream) {
    const float* x      = (const float*)d_in[0];
    const float* ln1_g  = (const float*)d_in[1];
    const float* ln1_b  = (const float*)d_in[2];
    const float* qkv_w  = (const float*)d_in[3];
    const float* qkv_b  = (const float*)d_in[4];
    const float* proj_w = (const float*)d_in[5];
    const float* proj_b = (const float*)d_in[6];
    const float* ln2_g  = (const float*)d_in[7];
    const float* ln2_b  = (const float*)d_in[8];
    const float* fc1_w  = (const float*)d_in[9];
    const float* fc1_b  = (const float*)d_in[10];
    const float* fc2_w  = (const float*)d_in[11];
    const float* fc2_b  = (const float*)d_in[12];
    float* out = (float*)d_out;

    char* w = (char*)d_ws;
    u16* wq  = (u16*)w; w += (size_t)CDIM * 3 * CDIM * 2;        // [3C, C] transposed
    u16* wp  = (u16*)w; w += (size_t)CDIM * CDIM * 2;            // [C, C] transposed
    u16* w1  = (u16*)w; w += (size_t)CDIM * HID * 2;             // [HID, C] transposed
    u16* w2  = (u16*)w; w += (size_t)HID * CDIM * 2;             // [C, HID] transposed
    u16* h   = (u16*)w; w += (size_t)NTOK * CDIM * 2;
    u16* qb  = (u16*)w; w += (size_t)NTOK * CDIM * 2;
    u16* kb  = (u16*)w; w += (size_t)NTOK * CDIM * 2;
    u16* vtb = (u16*)w; w += (size_t)NTOK * CDIM * 2;
    u16* ao  = (u16*)w; w += (size_t)NTOK * CDIM * 2;
    float* x2 = (float*)w; w += (size_t)NTOK * CDIM * 4;
    u16* hid = qb;  // alias: q/k/vt/ao region (25165824B) == hidden size, dead by fc1

    // 1) weight conversion + transpose to [N,K]
    cvt_t_kernel<<<512, 256, 0, stream>>>(qkv_w, wq, CDIM, 3 * CDIM);
    cvt_t_kernel<<<512, 256, 0, stream>>>(proj_w, wp, CDIM, CDIM);
    cvt_t_kernel<<<512, 256, 0, stream>>>(fc1_w, w1, CDIM, HID);
    cvt_t_kernel<<<512, 256, 0, stream>>>(fc2_w, w2, HID, CDIM);

    // 2) LN1
    ln_kernel<<<NTOK / 8, 256, 0, stream>>>(x, ln1_g, ln1_b, h);

    // 3) QKV gemm (M=8192, N=1152, K=384) with q/k/vt scatter (q pre-scaled)
    gemm_bf16<0><<<dim3(9, 64), 256, 0, stream>>>(h, wq, qkv_b, nullptr, nullptr,
                                                  nullptr, qb, kb, vtb,
                                                  NTOK, 3 * CDIM, CDIM);

    // 4) attention -> ao (bf16, [token, C])
    attn_kernel<<<384, 256, 0, stream>>>(qb, kb, vtb, ao);

    // 5) proj + residual -> x2 (fp32)
    gemm_bf16<1><<<dim3(3, 64), 256, 0, stream>>>(ao, wp, proj_b, x, x2,
                                                  nullptr, nullptr, nullptr, nullptr,
                                                  NTOK, CDIM, CDIM);

    // 6) LN2 -> h
    ln_kernel<<<NTOK / 8, 256, 0, stream>>>(x2, ln2_g, ln2_b, h);

    // 7) fc1 + GELU -> hid (bf16)
    gemm_bf16<2><<<dim3(12, 64), 256, 0, stream>>>(h, w1, fc1_b, nullptr, nullptr,
                                                   hid, nullptr, nullptr, nullptr,
                                                   NTOK, HID, CDIM);

    // 8) fc2 + residual -> d_out (fp32)
    gemm_bf16<1><<<dim3(3, 64), 256, 0, stream>>>(hid, w2, fc2_b, x2, out,
                                                  nullptr, nullptr, nullptr, nullptr,
                                                  NTOK, CDIM, HID);
}
